// CornerDetection_51367808860385
// MI455X (gfx1250) — matrix operand order repaired and checked by reading, not compiled, not run on hardware
//
#include <hip/hip_runtime.h>

// ---------------------------------------------------------------------------
// MI455X (gfx1250) fused Harris corner detection.
//   - TDM tensor_load_to_lds for bulk tile DMA (TENSORcnt / s_wait_tensorcnt)
//   - WMMA f32_16x16x32_f16 for all four separable conv stages (banded ops)
//   - Single pass over HBM: ~335 MB total -> ~14.4 us roofline @ 23.3 TB/s
// ---------------------------------------------------------------------------

typedef __attribute__((ext_vector_type(16))) _Float16 v16h;
typedef __attribute__((ext_vector_type(8)))  float    v8f;
typedef __attribute__((ext_vector_type(4)))  unsigned u32x4;
typedef __attribute__((ext_vector_type(8)))  int      i32x8;
typedef __attribute__((ext_vector_type(4)))  int      i32x4;

#ifndef __has_builtin
#define __has_builtin(x) 0
#endif
#if __has_builtin(__builtin_amdgcn_tensor_load_to_lds) && __has_builtin(__builtin_amdgcn_s_wait_tensorcnt)
#define HAVE_TDM 1
#else
#define HAVE_TDM 0
#endif

#define IMG    4096
#define TOUT   12            // valid output core per 16x16 tile (2 halo each side)
#define NTX    4             // tiles per block in x (one per wave column)
#define NTY    2
#define OW     (TOUT*NTX)    // 48 outputs / block in x
#define OH     (TOUT*NTY)    // 24 outputs / block in y
#define GW     (OW+4)        // 52 gray cols (halo 2)
#define GH     (OH+4)        // 28 gray rows
#define APITCH 18            // row pitch (halves) of A-layout staging

union H2 { unsigned u; _Float16 h[2]; };

__device__ __forceinline__ v8f wmma_f16(v16h a, v16h b, v8f c) {
  // (neg_a, A, neg_b, B, c_mod, C, reuse_a, reuse_b) -> v_wmma_f32_16x16x32_f16
  return __builtin_amdgcn_wmma_f32_16x16x32_f16(false, a, false, b, (short)0, c, false, false);
}

// Banded operator in A layout (16x32 f16; lane: M=lm; vgpr v: K = 2v + 16*lh).
template <class F>
__device__ __forceinline__ v16h opA(F f, int lm, int lh) {
  v16h a;
#pragma unroll
  for (int v = 0; v < 8; ++v) {
    int k = 2*v + lh*16;
    a[2*v]   = (_Float16)((k   < 16) ? f(lm, k  ) : 0.f);
    a[2*v+1] = (_Float16)((k+1 < 16) ? f(lm, k+1) : 0.f);
  }
  return a;
}
// Banded operator in B layout (32x16 f16; lane: N=lm; vgpr v: K = 2v + 16*lh).
template <class F>
__device__ __forceinline__ v16h opB(F f, int lm, int lh) {
  v16h b;
#pragma unroll
  for (int v = 0; v < 8; ++v) {
    int k = 2*v + lh*16;
    b[2*v]   = (_Float16)((k   < 16) ? f(k,   lm) : 0.f);
    b[2*v+1] = (_Float16)((k+1 < 16) ? f(k+1, lm) : 0.f);
  }
  return b;
}

// 16x16 data (row-major f16, pitch APITCH) loaded into A layout; K>=16 zero.
__device__ __forceinline__ v16h ldA(const _Float16* buf, int lm, int lh) {
  const unsigned* p = (const unsigned*)buf;
  v16h a;
#pragma unroll
  for (int v = 0; v < 8; ++v) {
    H2 t; t.u = lh ? 0u : p[(unsigned)(lm*APITCH + 2*v) >> 1];
    a[2*v] = t.h[0]; a[2*v+1] = t.h[1];
  }
  return a;
}
// 16x16 data (column-major f16, pitch 16 rows) loaded into B layout; K>=16 zero.
__device__ __forceinline__ v16h ldB16(const _Float16* buf, int lm, int lh) {
  v16h b;
  if (lh == 0) {
    const unsigned* p = (const unsigned*)buf;
#pragma unroll
    for (int v = 0; v < 8; ++v) {
      H2 t; t.u = p[(unsigned)(lm*16 + 2*v) >> 1];
      b[2*v] = t.h[0]; b[2*v+1] = t.h[1];
    }
  } else {
#pragma unroll
    for (int v = 0; v < 8; ++v) { b[2*v] = (_Float16)0.f; b[2*v+1] = (_Float16)0.f; }
  }
  return b;
}
// Gray tile (block-shared, column-major pitch GH) into B layout, origin (gr0,gc0).
__device__ __forceinline__ v16h ldB_gray(const _Float16* g, int gc0, int gr0, int lm, int lh) {
  v16h b;
  if (lh == 0) {
    const unsigned* p = (const unsigned*)g;
#pragma unroll
    for (int v = 0; v < 8; ++v) {
      H2 t; t.u = p[(unsigned)((gc0 + lm)*GH + gr0 + 2*v) >> 1];
      b[2*v] = t.h[0]; b[2*v+1] = t.h[1];
    }
  } else {
#pragma unroll
    for (int v = 0; v < 8; ++v) { b[2*v] = (_Float16)0.f; b[2*v+1] = (_Float16)0.f; }
  }
  return b;
}

__global__ __launch_bounds__(256) void harris_fused(const float* __restrict__ x,
                                                    float* __restrict__ out) {
  __shared__ float    s_raw[3*GH*GW];        // TDM destination: 3 clipped channel tiles
  __shared__ _Float16 s_gray[GW*GH];         // column-major [col*GH + row]
  __shared__ _Float16 s_A[8][3][16*APITCH];  // per-wave row-major staging (A operands)
  __shared__ _Float16 s_B[8][3][16*16];      // per-wave col-major products (B operands)

  const int tid = threadIdx.x;
  const int wv  = tid >> 5;
  const int ln  = tid & 31;
  const int lm  = ln & 15;       // M (A) / N (B,C,D)
  const int lh  = ln >> 4;       // lane half -> K group / +8 rows in C,D
  const int txi = wv & 3;
  const int tyi = wv >> 2;

  const int ox0 = blockIdx.x * OW;
  const int oy0 = blockIdx.y * OH;
  const int hx0 = ox0 - 2, hy0 = oy0 - 2;                 // logical halo origin
  const int cx0 = hx0 > 0 ? hx0 : 0;                      // clipped to image
  const int cy0 = hy0 > 0 ? hy0 : 0;
  const int cx1 = (hx0 + GW < IMG) ? (hx0 + GW) : IMG;
  const int cy1 = (hy0 + GH < IMG) ? (hy0 + GH) : IMG;
  const int w = cx1 - cx0, h = cy1 - cy0;                 // clipped tile dims

#if HAVE_TDM
  if (wv == 0) {
    const unsigned ldsb = (unsigned)(size_t)(void*)&s_raw[0];
    const unsigned planeBytes = (unsigned)(h * w) * 4u;
#pragma unroll
    for (int c = 0; c < 3; ++c) {
      unsigned long long ga =
          (unsigned long long)(size_t)(x + (size_t)c*IMG*IMG + (size_t)cy0*IMG + cx0);
      u32x4 g0;
      g0[0] = 1u;                                           // count=1, user mode
      g0[1] = ldsb + (unsigned)c * planeBytes;              // lds_addr
      g0[2] = (unsigned)(ga & 0xffffffffull);               // global_addr[31:0]
      g0[3] = (unsigned)((ga >> 32) & 0x01ffffffull) | (2u << 30); // addr[56:32] | type=2
      i32x8 g1;
      g1[0] = (int)(2u << 16);                              // data_size = 4B
      g1[1] = (int)(((unsigned)w & 0xffffu) << 16);         // tensor_dim0[15:0]
      g1[2] = (int)((((unsigned)w >> 16) & 0xffffu) | (((unsigned)h & 0xffffu) << 16));
      g1[3] = (int)((((unsigned)h >> 16) & 0xffffu) | (((unsigned)w & 0xffffu) << 16)); // tile_dim0 = w
      g1[4] = (int)((unsigned)h & 0xffffu);                 // tile_dim1 = h, tile_dim2 = 0
      g1[5] = (int)IMG;                                     // tensor_dim0_stride = 4096
      g1[6] = 0;                                            // stride0 hi | dim1_stride lo16 (=0)
      g1[7] = (int)(((unsigned)IMG * (unsigned)IMG) >> 16); // dim1_stride[47:16]
      i32x4 gz4 = {0, 0, 0, 0};
      i32x8 gz8 = {0, 0, 0, 0, 0, 0, 0, 0};
      // clang-23 / therock-10.0 arity: (g0, g1, g2, g3, extra, cpol)
      __builtin_amdgcn_tensor_load_to_lds(g0, g1, gz4, gz4, gz8, 0);
    }
    __builtin_amdgcn_s_wait_tensorcnt(0);
  }
#else
  for (int i = tid; i < 3*h*w; i += 256) {
    int c = i / (h*w); int rem = i - c*h*w;
    int ry = rem / w;  int rx = rem - ry*w;
    s_raw[i] = x[(size_t)c*IMG*IMG + (size_t)(cy0 + ry)*IMG + (cx0 + rx)];
  }
#endif
  __syncthreads();

  // Grayscale with zero padding outside the image; stored column-major f16.
  for (int i = tid; i < GW*GH; i += 256) {
    int lx = i % GW, ly = i / GW;
    int gx = hx0 + lx, gy = hy0 + ly;
    float g = 0.f;
    if (gx >= 0 && gx < IMG && gy >= 0 && gy < IMG) {
      int ri = (gy - cy0)*w + (gx - cx0);
      int pp = h*w;
      g = 0.2989f*s_raw[ri] + 0.587f*s_raw[pp + ri] + 0.114f*s_raw[2*pp + ri];
    }
    s_gray[lx*GH + ly] = (_Float16)g;
  }
  __syncthreads();

  const int gr0 = TOUT*tyi;       // tile gray origin inside the block region
  const int gc0 = TOUT*txi;
  const v8f zero8 = {0.f, 0.f, 0.f, 0.f, 0.f, 0.f, 0.f, 0.f};

  // --- Stage 1: vertical smooth + vertical diff (WMMA: D = Op x Gray) -------
  v16h Bg  = ldB_gray(s_gray, gc0, gr0, lm, lh);
  v16h A_s = opA([](int m, int k) { int d = k - m;
                   return d == 0 ? 0.625f : ((d == 1 || d == -1) ? 0.1875f : 0.f); }, lm, lh);
  v16h A_d = opA([](int m, int k) {
                   return k == m-1 ? 1.f : (k == m+1 ? -1.f : 0.f); }, lm, lh);
  v8f sv = wmma_f16(A_s, Bg, zero8);
  v8f dv = wmma_f16(A_d, Bg, zero8);

  _Float16* stSV = &s_A[wv][0][0];
  _Float16* stDV = &s_A[wv][1][0];
#pragma unroll
  for (int r = 0; r < 8; ++r) {
    stSV[(r + 8*lh)*APITCH + lm] = (_Float16)sv[r];
    stDV[(r + 8*lh)*APITCH + lm] = (_Float16)dv[r];
  }
  __syncthreads();

  // --- Stage 2: horizontal diff / smooth (WMMA: D = Data x Op) --------------
  v16h A_sv = ldA(stSV, lm, lh);
  v16h A_dv = ldA(stDV, lm, lh);
  v16h B_hd = opB([](int k, int n) {
                    return k == n-1 ? 1.f : (k == n+1 ? -1.f : 0.f); }, lm, lh);
  v16h B_hs = opB([](int k, int n) { int d = k - n;
                    return d == 0 ? 0.625f : ((d == 1 || d == -1) ? 0.1875f : 0.f); }, lm, lh);
  v8f ix = wmma_f16(A_sv, B_hd, zero8);
  v8f iy = wmma_f16(A_dv, B_hs, zero8);

  // Products (masked to 0 outside the image = box conv zero padding) + edge out
  float edge8[8];
  _Float16 pxx[8], pyy[8], pxy[8];
#pragma unroll
  for (int r = 0; r < 8; ++r) {
    int m = r + 8*lh, n = lm;
    int gxp = ox0 + gc0 + n - 2;
    int gyp = oy0 + gr0 + m - 2;
    bool in = (gxp >= 0) && (gxp < IMG) && (gyp >= 0) && (gyp < IMG);
    float fx = ix[r], fy = iy[r];
    edge8[r] = 0.5f*(fabsf(fx) + fabsf(fy));
    float msk = in ? 1.f : 0.f;
    pxx[r] = (_Float16)(fx*fx*msk);
    pyy[r] = (_Float16)(fy*fy*msk);
    pxy[r] = (_Float16)(fx*fy*msk);
  }
  {
    unsigned* qxx = (unsigned*)&s_B[wv][0][0];
    unsigned* qyy = (unsigned*)&s_B[wv][1][0];
    unsigned* qxy = (unsigned*)&s_B[wv][2][0];
#pragma unroll
    for (int r = 0; r < 8; r += 2) {
      unsigned idx = (unsigned)(lm*16 + r + 8*lh) >> 1;  // col-major, row pairs
      H2 t;
      t.h[0] = pxx[r]; t.h[1] = pxx[r+1]; qxx[idx] = t.u;
      t.h[0] = pyy[r]; t.h[1] = pyy[r+1]; qyy[idx] = t.u;
      t.h[0] = pxy[r]; t.h[1] = pxy[r+1]; qxy[idx] = t.u;
    }
  }
  __syncthreads();

  // --- Stage 3: vertical box sum (WMMA: D = Box x P) ------------------------
  v16h A_bv = opA([](int m, int k) { int d = k - m;
                    return (d >= -1 && d <= 1) ? 1.f : 0.f; }, lm, lh);
  v8f mvx = wmma_f16(A_bv, ldB16(&s_B[wv][0][0], lm, lh), zero8);
  v8f mvy = wmma_f16(A_bv, ldB16(&s_B[wv][1][0], lm, lh), zero8);
  v8f mvz = wmma_f16(A_bv, ldB16(&s_B[wv][2][0], lm, lh), zero8);

  _Float16* st0 = &s_A[wv][0][0];
  _Float16* st1 = &s_A[wv][1][0];
  _Float16* st2 = &s_A[wv][2][0];
#pragma unroll
  for (int r = 0; r < 8; ++r) {
    st0[(r + 8*lh)*APITCH + lm] = (_Float16)mvx[r];
    st1[(r + 8*lh)*APITCH + lm] = (_Float16)mvy[r];
    st2[(r + 8*lh)*APITCH + lm] = (_Float16)mvz[r];
  }
  __syncthreads();

  // --- Stage 4: horizontal box sum (WMMA: D = Mv x Box) ---------------------
  v16h B_bh = opB([](int k, int n) { int d = k - n;
                    return (d >= -1 && d <= 1) ? 1.f : 0.f; }, lm, lh);
  v8f Mxx = wmma_f16(ldA(st0, lm, lh), B_bh, zero8);
  v8f Myy = wmma_f16(ldA(st1, lm, lh), B_bh, zero8);
  v8f Mxy = wmma_f16(ldA(st2, lm, lh), B_bh, zero8);

  // --- Eigenvalue + edge outputs on the 12x12 valid core --------------------
#pragma unroll
  for (int r = 0; r < 8; ++r) {
    int m = r + 8*lh, n = lm;
    if (m < 2 || m >= 14 || n < 2 || n >= 14) continue;
    int gxp = ox0 + gc0 + n - 2;
    int gyp = oy0 + gr0 + m - 2;
    if (gxp >= IMG || gyp >= IMG) continue;
    float a = Mxx[r], b = Myy[r], c = Mxy[r];
    float diff = a - b, tr = a + b;
    float eig = tr - sqrtf(diff*diff + 4.f*c*c);
    size_t o = (size_t)gyp*IMG + gxp;
    out[o] = edge8[r];
    out[(size_t)IMG*IMG + o] = eig;
  }
}

extern "C" void kernel_launch(void* const* d_in, const int* in_sizes, int n_in,
                              void* d_out, int out_size, void* d_ws, size_t ws_size,
                              hipStream_t stream) {
  (void)in_sizes; (void)n_in; (void)d_ws; (void)ws_size; (void)out_size;
  const float* x  = (const float*)d_in[0];   // (1,3,4096,4096) f32; edge filter is constant
  float* out      = (float*)d_out;           // edge (4096^2) then eig (4096^2)
  dim3 grid((IMG + OW - 1)/OW, (IMG + OH - 1)/OH);
  harris_fused<<<grid, dim3(256), 0, stream>>>(x, out);
}
